// cross_set_score_20212116095167
// MI455X (gfx1250) — matrix operand order repaired and checked by reading, not compiled, not run on hardware
//
#include <hip/hip_runtime.h>
#include <hip/hip_bf16.h>

// ---------------------------------------------------------------------------
// cross_set_score for MI455X (gfx1250): bf16 WMMA pipeline
//   K1 proj : [x;y](65536x512) @ W(512x512) -> bf16, x stored transposed
//   K2 attn : per (i,j,h) 128x128x64 GEMM via v_wmma_f32_16x16x32_bf16,
//             tiles staged to LDS by the Tensor Data Mover (TDM)
//   K3 fin  : head combine with w2, size normalization
// ws layout: XPT (32MB bf16) | YP (32MB bf16) | SB (2048 f32)
// ---------------------------------------------------------------------------

typedef __attribute__((ext_vector_type(16))) __bf16 v16bf;
typedef __attribute__((ext_vector_type(8)))  __bf16 v8bf;
typedef __attribute__((ext_vector_type(8)))  float  v8f;
typedef __attribute__((ext_vector_type(4)))  unsigned int v4u;
typedef __attribute__((ext_vector_type(8)))  int    v8i;
typedef __attribute__((ext_vector_type(4)))  int    v4i;

#define NSET   16
#define NITEM  128
#define DIM    512
#define NHEAD  8
#define DHEAD  64

#if defined(__HIP_DEVICE_COMPILE__) && __has_builtin(__builtin_amdgcn_tensor_load_to_lds)
#define HAS_TDM 1
#else
#define HAS_TDM 0
#endif

static __device__ inline v16bf frag16(const __bf16* p0, const __bf16* p1) {
  v8bf lo = *(const v8bf*)p0;
  v8bf hi = *(const v8bf*)p1;
  return __builtin_shufflevector(lo, hi, 0,1,2,3,4,5,6,7,8,9,10,11,12,13,14,15);
}

static __device__ inline v8bf pack8(float4 a, float4 b) {
  v8bf r;
  r[0]=(__bf16)a.x; r[1]=(__bf16)a.y; r[2]=(__bf16)a.z; r[3]=(__bf16)a.w;
  r[4]=(__bf16)b.x; r[5]=(__bf16)b.y; r[6]=(__bf16)b.z; r[7]=(__bf16)b.w;
  return r;
}

// ---------------------------------------------------------------------------
// Kernel 1: projection GEMM.  Combined M = 2*16*16*128 = 65536 rows, N=512,
// K=512.  Block: 128x128 tile, 256 threads = 8 waves in 4(M) x 2(N).
// ---------------------------------------------------------------------------
__global__ void __launch_bounds__(256)
proj_kernel(const float* __restrict__ x, const float* __restrict__ y,
            const float* __restrict__ W,
            __bf16* __restrict__ XPT, __bf16* __restrict__ YP)
{
  const int nb  = blockIdx.x;           // 0..3   (N tile of 128)
  const int mb  = blockIdx.y;           // 0..511 (M tile of 128)
  const int tid = threadIdx.x;
  const int n0  = nb * 128;
  const long mrow0 = (long)mb * 128;
  const float* src = (mb < 256) ? x : y;
  const long  srow0 = (mb < 256) ? mrow0 : (mrow0 - 32768);

  __shared__ __align__(16) __bf16 lds_a[128 * 32];  // [m][k]
  __shared__ __align__(16) __bf16 lds_b[32 * 128];  // [k][n]

  const int wave = tid >> 5, lane = tid & 31;
  const int wm = wave & 3;              // 0..3 -> 32-row strip
  const int wn = wave >> 2;             // 0..1 -> 64-col strip
  const int lrow = lane & 15;
  const int kb8  = (lane >> 4) * 8;

  v8f c[2][4] = {};

  for (int kb = 0; kb < DIM; kb += 32) {
    // ---- stage A tile 128x32 (f32 -> bf16) ----
    {
      const int e0  = tid * 16;             // 16 elems / thread
      const int row = e0 >> 5;
      const int k0  = e0 & 31;              // 0 or 16
      const float4* gp = (const float4*)(src + (srow0 + row) * DIM + kb + k0);
      float4 f0 = gp[0], f1 = gp[1], f2 = gp[2], f3 = gp[3];
      *(v8bf*)&lds_a[row * 32 + k0]     = pack8(f0, f1);
      *(v8bf*)&lds_a[row * 32 + k0 + 8] = pack8(f2, f3);
    }
    // ---- stage B tile 32x128 of W (f32 -> bf16) ----
    {
      const int e0  = tid * 16;
      const int row = e0 >> 7;
      const int c0  = e0 & 127;
      const float4* gp = (const float4*)(W + (kb + row) * DIM + n0 + c0);
      float4 f0 = gp[0], f1 = gp[1], f2 = gp[2], f3 = gp[3];
      *(v8bf*)&lds_b[row * 128 + c0]     = pack8(f0, f1);
      *(v8bf*)&lds_b[row * 128 + c0 + 8] = pack8(f2, f3);
    }
    __syncthreads();

    // ---- fragments + WMMA ----
    v16bf af[2];
#pragma unroll
    for (int tm = 0; tm < 2; tm++) {
      const __bf16* p = &lds_a[(wm * 32 + tm * 16 + lrow) * 32 + kb8];
      af[tm] = frag16(p, p + 16);
    }
    v16bf bfm[4];
#pragma unroll
    for (int tn = 0; tn < 4; tn++) {
      const __bf16* p = &lds_b[kb8 * 128 + wn * 64 + tn * 16 + lrow];
#pragma unroll
      for (int s = 0; s < 16; s++) bfm[tn][s] = p[(s < 8 ? s : s + 8) * 128];
    }
#pragma unroll
    for (int tm = 0; tm < 2; tm++)
#pragma unroll
      for (int tn = 0; tn < 4; tn++)
        c[tm][tn] = __builtin_amdgcn_wmma_f32_16x16x32_bf16(
            false, af[tm], false, bfm[tn], (short)0, c[tm][tn], false, false);
    __syncthreads();
  }

  // ---- store results ----
  // C/D layout: VGPR v, lanes 0-15 -> M=v, N=lane; lanes 16-31 -> M=v+8.
  const int lhi = lane >> 4;
#pragma unroll
  for (int tm = 0; tm < 2; tm++) {
#pragma unroll
    for (int tn = 0; tn < 4; tn++) {
      const int mloc = wm * 32 + tm * 16 + lhi * 8;   // +v for v=0..7
      const int gc   = n0 + wn * 64 + tn * 16 + lrow; // fixed column
      if (mb < 256) {
        // x rows: pair p == mb, a = mloc+v.  XPT[p][h][d][a] (a contiguous)
        const int h = gc >> 6, d = gc & 63;
        v8bf pk;
#pragma unroll
        for (int v = 0; v < 8; v++) pk[v] = (__bf16)c[tm][tn][v];
        __bf16* dst = XPT + ((((long)mb * NHEAD + h) * DHEAD + d) * NITEM + mloc);
        *(v8bf*)dst = pk;
      } else {
        // y rows: natural row-major YP[row][hd]
        const long ry = mrow0 - 32768 + mloc;
#pragma unroll
        for (int v = 0; v < 8; v++)
          YP[(ry + v) * DIM + gc] = (__bf16)c[tm][tn][v];
      }
    }
  }
}

// ---------------------------------------------------------------------------
// TDM: build a 2D tensor descriptor (D#) and issue tensor_load_to_lds.
// dims/strides in elements of 2 bytes (data_size code 1).
// This toolchain exposes the 6-arg builtin:
//   (uint32x4 g0, int32x8 g1, int32x4, int32x4, int32x8, i32 cpol)
// Groups 2/3 (and the trailing group) are zero = unused (<=2D tensor).
// ---------------------------------------------------------------------------
#if HAS_TDM
static __device__ inline void tdm_load_2d(unsigned lds_off, const void* gaddr,
                                          unsigned tile_d0, unsigned tile_d1,
                                          unsigned tensor_d0, unsigned tensor_d1,
                                          unsigned stride0)
{
  unsigned long long ga = (unsigned long long)gaddr;
  v4u g0;
  g0[0] = 1u;                                              // count=1 (user D#)
  g0[1] = lds_off;                                         // lds_addr (bytes)
  g0[2] = (unsigned)(ga & 0xFFFFFFFFu);                    // global_addr lo
  g0[3] = (unsigned)((ga >> 32) & 0x1FFFFFFu) | (2u << 30);// addr hi | type=2
  v8i g1;
  g1[0] = (int)(1u << 16);                                 // data_size=2B
  g1[1] = (int)((tensor_d0 & 0xFFFFu) << 16);              // dim0[15:0]
  g1[2] = (int)(((tensor_d0 >> 16) & 0xFFFFu) |
                ((tensor_d1 & 0xFFFFu) << 16));            // dim0[31:16]|dim1 lo
  g1[3] = (int)(((tensor_d1 >> 16) & 0xFFFFu) |
                ((tile_d0 & 0xFFFFu) << 16));              // dim1 hi | tile_dim0
  g1[4] = (int)(tile_d1 & 0xFFFFu);                        // tile_dim1, tile_dim2=0
  g1[5] = (int)stride0;                                    // dim0_stride lo
  g1[6] = 0;
  g1[7] = 0;
  v4i z4 = {0, 0, 0, 0};
  v8i z8 = {0, 0, 0, 0, 0, 0, 0, 0};
  __builtin_amdgcn_tensor_load_to_lds(g0, g1, z4, z4, z8, 0);
}
#endif

// ---------------------------------------------------------------------------
// Kernel 2: per (i,j,h) attention block.  A = yp (128x64, m=b,k=d),
// B = xpT (64x128, k=d,n=a).  Accumulate relu-sum -> SB[(i*16+j)*8+h].
// ---------------------------------------------------------------------------
__global__ void __launch_bounds__(256)
attn_kernel(const __bf16* __restrict__ XPT, const __bf16* __restrict__ YP,
            float* __restrict__ SB)
{
  const int h = blockIdx.x, j = blockIdx.y, i = blockIdx.z;
  const int px = i * NSET + j;
  const int py = j * NSET + i;
  const int tid = threadIdx.x, wave = tid >> 5, lane = tid & 31;
  const int wm = wave & 3, wn = wave >> 2;
  const int lrow = lane & 15, kb8 = (lane >> 4) * 8;

  extern __shared__ __align__(16) __bf16 smem[];
  __bf16* lds_y = smem;                 // [128][64]  b x d
  __bf16* lds_x = smem + 128 * 64;      // [64][128]  d x a
  float*  red   = (float*)(smem + 128 * 64 + 64 * 128);

  const __bf16* gy = YP + ((long)py * NITEM) * DIM + h * DHEAD;
  const __bf16* gx = XPT + (((long)px * NHEAD + h) * DHEAD) * NITEM;

#if HAS_TDM
  if (tid < 32) {   // wave 0 issues both DMA descriptors
    tdm_load_2d((unsigned)(unsigned long long)lds_y, gy,
                /*tile*/ 64, 128, /*tensor*/ 64, 128, /*stride0*/ DIM);
    tdm_load_2d((unsigned)(unsigned long long)lds_x, gx,
                /*tile*/ 128, 64, /*tensor*/ 128, 64, /*stride0*/ NITEM);
    __builtin_amdgcn_s_wait_tensorcnt(0);
  }
#else
  for (int e = tid * 8; e < 128 * 64; e += 256 * 8) {
    const int row = e >> 6, col = e & 63;
    *(v8bf*)&lds_y[e] = *(const v8bf*)&gy[(long)row * DIM + col];
  }
  for (int e = tid * 8; e < 64 * 128; e += 256 * 8)
    *(v8bf*)&lds_x[e] = *(const v8bf*)&gx[e];
#endif
  if (tid == 0) *red = 0.0f;
  __syncthreads();

  v8f c[2][4] = {};
#pragma unroll
  for (int k0 = 0; k0 < DHEAD; k0 += 32) {
    v16bf af[2];
#pragma unroll
    for (int tm = 0; tm < 2; tm++) {
      const __bf16* p = &lds_y[(wm * 32 + tm * 16 + lrow) * DHEAD + k0 + kb8];
      af[tm] = frag16(p, p + 16);
    }
    v16bf bfm[4];
#pragma unroll
    for (int tn = 0; tn < 4; tn++) {
      const __bf16* p = &lds_x[(k0 + kb8) * NITEM + wn * 64 + tn * 16 + lrow];
#pragma unroll
      for (int s = 0; s < 16; s++) bfm[tn][s] = p[(s < 8 ? s : s + 8) * NITEM];
    }
#pragma unroll
    for (int tm = 0; tm < 2; tm++)
#pragma unroll
      for (int tn = 0; tn < 4; tn++)
        c[tm][tn] = __builtin_amdgcn_wmma_f32_16x16x32_bf16(
            false, af[tm], false, bfm[tn], (short)0, c[tm][tn], false, false);
  }

  // relu + full reduction
  float s = 0.0f;
#pragma unroll
  for (int tm = 0; tm < 2; tm++)
#pragma unroll
    for (int tn = 0; tn < 4; tn++)
#pragma unroll
      for (int v = 0; v < 8; v++)
        s += fmaxf(c[tm][tn][v], 0.0f);
#pragma unroll
  for (int off = 16; off > 0; off >>= 1)
    s += __shfl_xor(s, off, 32);
  if ((tid & 31) == 0) atomicAdd(red, s);
  __syncthreads();
  if (tid == 0) SB[(long)px * NHEAD + h] = *red;
}

// ---------------------------------------------------------------------------
// Kernel 3: combine heads.  scores[j,i] = sum_h SB[i,j,h]*w2[h] / (8*xs*ys)
// (relu(a/8) = relu(a)/8, so the 1/sqrt(Dh) scale commutes out of the sum.)
// ---------------------------------------------------------------------------
__global__ void __launch_bounds__(256)
final_kernel(const float* __restrict__ SB, const float* __restrict__ x_size,
             const float* __restrict__ y_size, const float* __restrict__ w2,
             float* __restrict__ out)
{
  const int t = threadIdx.x;
  const int j = t >> 4, i = t & 15;
  float acc = 0.0f;
#pragma unroll
  for (int h = 0; h < NHEAD; h++)
    acc += SB[(i * NSET + j) * NHEAD + h] * w2[h];
  out[j * NSET + i] = acc / (8.0f * x_size[i] * y_size[j]);
}

// ---------------------------------------------------------------------------
extern "C" void kernel_launch(void* const* d_in, const int* in_sizes, int n_in,
                              void* d_out, int out_size, void* d_ws, size_t ws_size,
                              hipStream_t stream)
{
  (void)in_sizes; (void)n_in; (void)out_size; (void)ws_size;
  const float* x  = (const float*)d_in[0];
  const float* y  = (const float*)d_in[1];
  const float* xs = (const float*)d_in[2];
  const float* ys = (const float*)d_in[3];
  const float* W  = (const float*)d_in[4];
  const float* w2 = (const float*)d_in[5];
  float* out = (float*)d_out;

  const size_t XPT_ELEMS = (size_t)NSET * NSET * NHEAD * DHEAD * NITEM; // 16M
  const size_t YP_ELEMS  = (size_t)NSET * NSET * NITEM * DIM;           // 16M
  __bf16* XPT = (__bf16*)d_ws;
  __bf16* YP  = XPT + XPT_ELEMS;
  float*  SB  = (float*)(YP + YP_ELEMS);

  proj_kernel<<<dim3(4, 512, 1), 256, 0, stream>>>(x, y, W, XPT, YP);
  attn_kernel<<<dim3(NHEAD, NSET, NSET), 256,
                (128 * 64 + 64 * 128) * sizeof(__bf16) + 16, stream>>>(XPT, YP, SB);
  final_kernel<<<1, 256, 0, stream>>>(SB, xs, ys, w2, out);
}
